// DGCNN_Seg_Encoder_40587440947664
// MI455X (gfx1250) — matrix operand order repaired and checked by reading, not compiled, not run on hardware
//
#include <hip/hip_runtime.h>
#include <hip/hip_bf16.h>
#include <cstdint>

// ---------------------------------------------------------------------------
// DGCNN segmentation encoder for MI455X (gfx1250), wave32 + WMMA bf16 + TDM.
// B=8, N=2048, K=20, F=1024.  GEMMs via v_wmma_f32_16x16x32_bf16; KNN point
// cache loaded with tensor_load_to_lds (TENSORcnt).  All weights are
// converted once to K-padded bf16 in workspace so B fragments are pure
// 16B loads (no per-tile cvt).
// ---------------------------------------------------------------------------

#define BATCH 8
#define NPTS  2048
#define KNNK  20
#define EPS_  1e-5f
#define LRELU 0.2f

typedef __bf16 bf16_t;
typedef __attribute__((ext_vector_type(16))) __bf16 bf16x16;
typedef __attribute__((ext_vector_type(8)))  __bf16 bf16x8;
typedef __attribute__((ext_vector_type(4)))  __bf16 bf16x4;
typedef __attribute__((ext_vector_type(8)))  float  f32x8;
typedef __attribute__((ext_vector_type(4)))  unsigned int u32x4;
typedef __attribute__((ext_vector_type(8)))  int i32x8;
typedef __attribute__((ext_vector_type(4)))  int i32x4;

// ------------------------- WMMA fragment loaders ---------------------------
// A matrix 16x32 bf16 (MxK) from LDS, row-major, stride ld elements.
// Lanes 0-15 = rows with K 0..15; lanes 16-31 same rows,
// K 16..31.  Per lane: two contiguous 16B runs.
__device__ __forceinline__ bf16x16 load_a_lds(const bf16_t* base, int ld) {
  const int lane = threadIdx.x & 31;
  const int row  = lane & 15;
  const int kb   = (lane < 16) ? 0 : 16;
  const bf16_t* p = base + row * ld + kb;
  bf16x8 lo = *(const bf16x8*)(p);
  bf16x8 hi = *(const bf16x8*)(p + 8);
  return __builtin_shufflevector(lo, hi, 0, 1, 2, 3, 4, 5, 6, 7,
                                 8, 9, 10, 11, 12, 13, 14, 15);
}

// B matrix 32x16 bf16 (KxN) from pre-converted weights Wb (rows x ldk bf16,
// K-padded, row-major over outputs).  B[k][n] = Wb[o0+n][c0+k].
// Lanes 0-15: K 0..15; lanes 16-31: K 16..31 -> one 32B run per lane.
__device__ __forceinline__ bf16x16 load_b_frag(const bf16_t* __restrict__ Wb,
                                               int ldk, int o0, int c0) {
  const int lane = threadIdx.x & 31;
  const int n    = lane & 15;
  const int kb   = (lane < 16) ? 0 : 16;
  const bf16_t* p = Wb + (size_t)(o0 + n) * ldk + c0 + kb;
  bf16x8 lo = *(const bf16x8*)(p);
  bf16x8 hi = *(const bf16x8*)(p + 8);
  return __builtin_shufflevector(lo, hi, 0, 1, 2, 3, 4, 5, 6, 7,
                                 8, 9, 10, 11, 12, 13, 14, 15);
}

// -------------------- weight f32 -> K-padded bf16 prep ---------------------
__global__ __launch_bounds__(256) void wcvt_kernel(const float* __restrict__ W,
                                                   bf16_t* __restrict__ out,
                                                   int kin, int kpad, int rows) {
  const int t = blockIdx.x * 256 + threadIdx.x;
  if (t >= rows * kpad) return;
  const int o = t / kpad, k = t - o * kpad;
  out[t] = (k < kin) ? (bf16_t)W[(size_t)o * kin + k] : (bf16_t)0.f;
}

// ------------------------------- KNN (3-dim) -------------------------------
__global__ __launch_bounds__(256) void knn3_kernel(const float* __restrict__ src,
                                                   int stride, int offset,
                                                   int* __restrict__ idx_out) {
  __shared__ float pts[NPTS * 3];
  const int b = blockIdx.y;
  const int q = blockIdx.x * 256 + threadIdx.x;
  const float* base = src + (size_t)b * NPTS * stride + offset;

#if __has_builtin(__builtin_amdgcn_tensor_load_to_lds)
  // Tensor Data Mover: DMA the 2048x3 f32 tile (row stride = `stride`) into
  // LDS.  D# packing per CDNA5 ISA.
  {
    const uint64_t ga      = (uint64_t)(uintptr_t)base;
    const uint32_t lds_off = (uint32_t)(uintptr_t)(void*)pts;  // addr[31:0] = LDS offset
    u32x4 g0 = { 1u,                                             // count=1 (valid user D#)
                 lds_off,                                        // lds_addr
                 (uint32_t)ga,                                   // global_addr[31:0]
                 (uint32_t)((ga >> 32) & 0x01FFFFFFu) | (2u << 30) }; // addr[56:32] | type=2
    i32x8 g1 = { (int)(2u << 16),          // data_size = 4 bytes
                 (int)(3u << 16),          // tensor_dim0 = 3
                 (int)(2048u << 16),       // tensor_dim1 = 2048
                 (int)(3u << 16),          // tile_dim0 = 3
                 (int)2048,                // tile_dim1 = 2048 (tile_dim2 = 0)
                 (int)stride,              // tensor_dim0_stride (elements)
                 0, 0 };
    i32x4 gz = { 0, 0, 0, 0 };
    if (threadIdx.x < 32) {                // one wave issues the DMA
#if defined(__clang_major__) && (__clang_major__ >= 23)
      i32x8 gz8 = { 0, 0, 0, 0, 0, 0, 0, 0 };
      __builtin_amdgcn_tensor_load_to_lds(g0, g1, gz, gz, gz8, 0);
#else
      __builtin_amdgcn_tensor_load_to_lds(g0, g1, gz, gz, 0);
#endif
#if __has_builtin(__builtin_amdgcn_s_wait_tensorcnt)
      __builtin_amdgcn_s_wait_tensorcnt(0);
#endif
    }
  }
#else
  for (int i = threadIdx.x; i < NPTS; i += 256) {
    pts[i * 3 + 0] = base[(size_t)i * stride + 0];
    pts[i * 3 + 1] = base[(size_t)i * stride + 1];
    pts[i * 3 + 2] = base[(size_t)i * stride + 2];
  }
#endif
  __syncthreads();

  const float qx = pts[q * 3 + 0], qy = pts[q * 3 + 1], qz = pts[q * 3 + 2];
  const float sqq = qx * qx + qy * qy + qz * qz;

  float vals[KNNK]; int ids[KNNK];
#pragma unroll
  for (int i = 0; i < KNNK; ++i) { vals[i] = -3.0e38f; ids[i] = 0; }

  for (int m = 0; m < NPTS; ++m) {
    float dx = pts[m * 3 + 0], dy = pts[m * 3 + 1], dz = pts[m * 3 + 2];
    float nd = 2.f * (qx * dx + qy * dy + qz * dz) - sqq -
               (dx * dx + dy * dy + dz * dz);
    if (nd > vals[KNNK - 1]) {
#pragma unroll
      for (int i = KNNK - 1; i > 0; --i) {
        bool shift = (nd > vals[i - 1]);
        if (shift)             { vals[i] = vals[i - 1]; ids[i] = ids[i - 1]; }
        else if (nd > vals[i]) { vals[i] = nd;          ids[i] = m;          }
      }
      if (nd > vals[0]) { vals[0] = nd; ids[0] = m; }
    }
  }
  int* o = idx_out + ((size_t)b * NPTS + q) * KNNK;
#pragma unroll
  for (int i = 0; i < KNNK; ++i) o[i] = ids[i];
}

// ---------------------- fused edge-conv GEMM layer -------------------------
// 320 rows x 64 outs = 20 M-tiles x 4 N-tiles; 8 waves x 10 tiles.
template<int KDIM>
__device__ __forceinline__ void gemm_bn_layer(const bf16_t* __restrict__ Ain, int lda,
                                              const bf16_t* __restrict__ Wb,
                                              const float* __restrict__ gg,
                                              const float* __restrict__ bb,
                                              const float* __restrict__ mm,
                                              const float* __restrict__ vv,
                                              bf16_t* outBf, float* outF) {
  const int wave  = threadIdx.x >> 5;
  const int lane  = threadIdx.x & 31;
  const int nl    = lane & 15;
  const int mbase = (lane < 16) ? 0 : 8;
  for (int t = wave; t < 20 * 4; t += 8) {
    const int mt = t >> 2, nt = t & 3;
    f32x8 acc = {};
#pragma unroll
    for (int ks = 0; ks < KDIM / 32; ++ks) {
      bf16x16 a  = load_a_lds(Ain + (size_t)(mt * 16) * lda + ks * 32, lda);
      bf16x16 bf = load_b_frag(Wb, KDIM, nt * 16, ks * 32);
      acc = __builtin_amdgcn_wmma_f32_16x16x32_bf16(false, a, false, bf,
                                                    (short)0, acc, false, false);
    }
#pragma unroll
    for (int r = 0; r < 8; ++r) {
      const int m   = mbase + r;
      const int o   = nt * 16 + nl;
      const int row = mt * 16 + m;
      float z = acc[r];
      z = (z - mm[o]) * rsqrtf(vv[o] + EPS_) * gg[o] + bb[o];
      z = (z >= 0.f) ? z : LRELU * z;
      if (outBf) outBf[(size_t)row * 64 + o] = (bf16_t)z;
      else       outF [(size_t)row * 64 + o] = z;
    }
  }
}

// IN_CH = edge-feature dim (6 or 128); padded to 32 for WMMA K.
template<int IN_CH, bool TWO_LAYER>
__global__ __launch_bounds__(256) void edgeconv_kernel(
    const float* __restrict__ xin, const int* __restrict__ knn_idx,
    const bf16_t* __restrict__ Wb1,
    const float* __restrict__ g1, const float* __restrict__ b1,
    const float* __restrict__ m1, const float* __restrict__ v1,
    const bf16_t* __restrict__ Wb2,
    const float* __restrict__ g2, const float* __restrict__ b2,
    const float* __restrict__ m2, const float* __restrict__ v2,
    float* __restrict__ xout) {
  constexpr int CIN    = IN_CH / 2;
  constexpr int IN_PAD = (IN_CH <= 32) ? 32 : IN_CH;
  constexpr int ROWS   = 16 * KNNK;          // 320

  extern __shared__ char smem[];
  char* region0 = smem;                               // E (bf16) / Hout (f32) alias
  char* region1 = smem + (size_t)ROWS * 64 * 4;       // H1 bf16 or Hout f32
  bf16_t* E    = (bf16_t*)region0;                    // ROWS x IN_PAD
  bf16_t* H1   = (bf16_t*)region1;                    // ROWS x 64
  float*  Hout = TWO_LAYER ? (float*)region0 : (float*)region1;

  const int b  = blockIdx.y;
  const int n0 = blockIdx.x * 16;
  const float* xb = xin + (size_t)b * NPTS * CIN;

  // Build edge features: [nb - ctr, ctr], zero-padded to IN_PAD.
  for (int r = threadIdx.x; r < ROWS; r += 256) {
    const int p = r / KNNK, j = r - p * KNNK;
    const int n = n0 + p;
    const int nb = knn_idx[((size_t)b * NPTS + n) * KNNK + j];
    const float* ctr = xb + (size_t)n  * CIN;
    const float* nbp = xb + (size_t)nb * CIN;
    bf16_t* er = E + (size_t)r * IN_PAD;
    if constexpr (CIN == 64) {
      const float4* c4 = (const float4*)ctr;
      const float4* n4 = (const float4*)nbp;
      bf16x4* er4 = (bf16x4*)er;
#pragma unroll
      for (int i = 0; i < 16; ++i) {
        const float4 cv = c4[i], nv = n4[i];
        bf16x4 d, c;
        d[0] = (bf16_t)(nv.x - cv.x); d[1] = (bf16_t)(nv.y - cv.y);
        d[2] = (bf16_t)(nv.z - cv.z); d[3] = (bf16_t)(nv.w - cv.w);
        c[0] = (bf16_t)cv.x; c[1] = (bf16_t)cv.y;
        c[2] = (bf16_t)cv.z; c[3] = (bf16_t)cv.w;
        er4[i]      = d;
        er4[16 + i] = c;
      }
    } else {
#pragma unroll
      for (int c = 0; c < CIN; ++c) {
        const float cv = ctr[c];
        er[c]       = (bf16_t)(nbp[c] - cv);
        er[CIN + c] = (bf16_t)cv;
      }
#pragma unroll
      for (int c = 2 * CIN; c < IN_PAD; ++c) er[c] = (bf16_t)0.f;
    }
  }
  __syncthreads();

  if constexpr (TWO_LAYER) {
    gemm_bn_layer<IN_PAD>(E, IN_PAD, Wb1, g1, b1, m1, v1, H1, nullptr);
    __syncthreads();
    gemm_bn_layer<64>(H1, 64, Wb2, g2, b2, m2, v2, nullptr, Hout);
  } else {
    (void)Wb2; (void)g2; (void)b2; (void)m2; (void)v2;
    gemm_bn_layer<IN_PAD>(E, IN_PAD, Wb1, g1, b1, m1, v1, nullptr, Hout);
  }
  __syncthreads();

  // Max over the 20 neighbors, write (B,N,64).
  float* xo = xout + ((size_t)b * NPTS + n0) * 64;
  for (int t = threadIdx.x; t < 16 * 64; t += 256) {
    const int p = t >> 6, c = t & 63;
    float mx = -3.0e38f;
#pragma unroll
    for (int j = 0; j < KNNK; ++j)
      mx = fmaxf(mx, Hout[(size_t)(p * KNNK + j) * 64 + c]);
    xo[(size_t)p * 64 + c] = mx;
  }
}

// ------------------- final FC (192->1024) + partial max --------------------
__global__ __launch_bounds__(256) void fc_max_kernel(
    const float* __restrict__ x1, const float* __restrict__ x2,
    const float* __restrict__ x3, const bf16_t* __restrict__ W6b,
    const float* __restrict__ g6, const float* __restrict__ b6,
    const float* __restrict__ m6, const float* __restrict__ v6,
    float* __restrict__ x4t, float* __restrict__ part) {
  extern __shared__ char smem[];
  bf16_t* X4 = (bf16_t*)smem;                       // 16 x 192
  float*  ZL = (float*)(smem + 16 * 192 * 2);       // 16 x 1024

  const int b  = blockIdx.y;
  const int n0 = blockIdx.x * 16;

  // Build x4 tile (concat x1|x2|x3) and emit transposed output (B,192,N).
  for (int t = threadIdx.x; t < 16 * 192; t += 256) {
    const int p = t / 192, c = t - p * 192;
    const int n = n0 + p;
    float f;
    if      (c < 64)  f = x1[((size_t)b * NPTS + n) * 64 + c];
    else if (c < 128) f = x2[((size_t)b * NPTS + n) * 64 + (c - 64)];
    else              f = x3[((size_t)b * NPTS + n) * 64 + (c - 128)];
    X4[p * 192 + c] = (bf16_t)f;
    x4t[((size_t)b * 192 + c) * NPTS + n] = f;
  }
  __syncthreads();

  const int wave  = threadIdx.x >> 5;
  const int lane  = threadIdx.x & 31;
  const int nl    = lane & 15;
  const int mbase = (lane < 16) ? 0 : 8;
  for (int nt = wave; nt < 64; nt += 8) {
    f32x8 acc = {};
#pragma unroll
    for (int ks = 0; ks < 6; ++ks) {
      bf16x16 a  = load_a_lds(X4 + ks * 32, 192);
      bf16x16 bf = load_b_frag(W6b, 192, nt * 16, ks * 32);
      acc = __builtin_amdgcn_wmma_f32_16x16x32_bf16(false, a, false, bf,
                                                    (short)0, acc, false, false);
    }
#pragma unroll
    for (int r = 0; r < 8; ++r) {
      const int m = mbase + r;
      const int o = nt * 16 + nl;
      float z = acc[r];
      z = (z - m6[o]) * rsqrtf(v6[o] + EPS_) * g6[o] + b6[o];
      z = (z >= 0.f) ? z : LRELU * z;
      ZL[m * 1024 + o] = z;
    }
  }
  __syncthreads();

  for (int c = threadIdx.x; c < 1024; c += 256) {
    float mx = -3.0e38f;
#pragma unroll
    for (int p = 0; p < 16; ++p) mx = fmaxf(mx, ZL[p * 1024 + c]);
    part[(((size_t)b * (NPTS / 16)) + blockIdx.x) * 1024 + c] = mx;
  }
}

__global__ __launch_bounds__(256) void gmax_kernel(const float* __restrict__ part,
                                                   float* __restrict__ feat) {
  const int b = blockIdx.y;
  const int c = blockIdx.x * 256 + threadIdx.x;
  float mx = -3.0e38f;
  for (int t = 0; t < NPTS / 16; ++t)
    mx = fmaxf(mx, part[(((size_t)b * (NPTS / 16)) + t) * 1024 + c]);
  feat[(size_t)b * 1024 + c] = mx;
}

// ---------------------------------------------------------------------------
extern "C" void kernel_launch(void* const* d_in, const int* in_sizes, int n_in,
                              void* d_out, int out_size, void* d_ws, size_t ws_size,
                              hipStream_t stream) {
  (void)in_sizes; (void)n_in; (void)out_size; (void)ws_size;
  const float* x   = (const float*)d_in[0];
  const float* W1  = (const float*)d_in[1];
  const float* W2  = (const float*)d_in[2];
  const float* W3  = (const float*)d_in[3];
  const float* W4  = (const float*)d_in[4];
  const float* W5  = (const float*)d_in[5];
  const float* W6  = (const float*)d_in[6];
  const float* bng = (const float*)d_in[7];
  const float* bnb = (const float*)d_in[8];
  const float* bnm = (const float*)d_in[9];
  const float* bnv = (const float*)d_in[10];
  const float* g6  = (const float*)d_in[11];
  const float* b6  = (const float*)d_in[12];
  const float* m6  = (const float*)d_in[13];
  const float* v6  = (const float*)d_in[14];

  char* ws = (char*)d_ws;
  size_t off = 0;
  int*    idx  = (int*)(ws + off);    off += (size_t)BATCH * NPTS * KNNK * sizeof(int);
  float*  x1   = (float*)(ws + off);  off += (size_t)BATCH * NPTS * 64 * sizeof(float);
  float*  x2   = (float*)(ws + off);  off += (size_t)BATCH * NPTS * 64 * sizeof(float);
  float*  x3   = (float*)(ws + off);  off += (size_t)BATCH * NPTS * 64 * sizeof(float);
  float*  part = (float*)(ws + off);  off += (size_t)BATCH * (NPTS / 16) * 1024 * sizeof(float);
  bf16_t* Wb1  = (bf16_t*)(ws + off); off += (size_t)64 * 32 * sizeof(bf16_t);
  bf16_t* Wb2  = (bf16_t*)(ws + off); off += (size_t)64 * 64 * sizeof(bf16_t);
  bf16_t* Wb3  = (bf16_t*)(ws + off); off += (size_t)64 * 128 * sizeof(bf16_t);
  bf16_t* Wb4  = (bf16_t*)(ws + off); off += (size_t)64 * 64 * sizeof(bf16_t);
  bf16_t* Wb5  = (bf16_t*)(ws + off); off += (size_t)64 * 128 * sizeof(bf16_t);
  bf16_t* Wb6  = (bf16_t*)(ws + off); off += (size_t)1024 * 192 * sizeof(bf16_t);

  float* feat = (float*)d_out;                        // (B,1,1024)
  float* x4t  = (float*)d_out + (size_t)BATCH * 1024; // (B,192,N)

  const dim3 blk(256);
  const dim3 grd_knn(NPTS / 256, BATCH);
  const dim3 grd_ec(NPTS / 16, BATCH);
  const size_t lds_ec = 2 * (size_t)(320 * 64 * 4);        // 160 KB (E/Hout + H1)
  const size_t lds_fc = 16 * 192 * 2 + 16 * 1024 * 4;      // 70 KB

  // One-time weight conversion to K-padded bf16 (L2-resident thereafter).
  wcvt_kernel<<<(64 * 32 + 255) / 256, blk, 0, stream>>>(W1, Wb1, 6, 32, 64);
  wcvt_kernel<<<(64 * 64 + 255) / 256, blk, 0, stream>>>(W2, Wb2, 64, 64, 64);
  wcvt_kernel<<<(64 * 128 + 255) / 256, blk, 0, stream>>>(W3, Wb3, 128, 128, 64);
  wcvt_kernel<<<(64 * 64 + 255) / 256, blk, 0, stream>>>(W4, Wb4, 64, 64, 64);
  wcvt_kernel<<<(64 * 128 + 255) / 256, blk, 0, stream>>>(W5, Wb5, 128, 128, 64);
  wcvt_kernel<<<(1024 * 192 + 255) / 256, blk, 0, stream>>>(W6, Wb6, 192, 192, 1024);

  // Stage 1
  knn3_kernel<<<grd_knn, blk, 0, stream>>>(x, 3, 0, idx);
  edgeconv_kernel<6, true><<<grd_ec, blk, lds_ec, stream>>>(
      x, idx, Wb1, bng + 0, bnb + 0, bnm + 0, bnv + 0,
      Wb2, bng + 64, bnb + 64, bnm + 64, bnv + 64, x1);
  // Stage 2 (KNN on channels 6:9 of x1)
  knn3_kernel<<<grd_knn, blk, 0, stream>>>(x1, 64, 6, idx);
  edgeconv_kernel<128, true><<<grd_ec, blk, lds_ec, stream>>>(
      x1, idx, Wb3, bng + 128, bnb + 128, bnm + 128, bnv + 128,
      Wb4, bng + 192, bnb + 192, bnm + 192, bnv + 192, x2);
  // Stage 3 (single layer)
  knn3_kernel<<<grd_knn, blk, 0, stream>>>(x2, 64, 6, idx);
  edgeconv_kernel<128, false><<<grd_ec, blk, lds_ec, stream>>>(
      x2, idx, Wb5, bng + 256, bnb + 256, bnm + 256, bnv + 256,
      nullptr, nullptr, nullptr, nullptr, nullptr, x3);
  // Final FC + global max
  fc_max_kernel<<<grd_ec, blk, lds_fc, stream>>>(x1, x2, x3, Wb6, g6, b6, m6, v6,
                                                 x4t, part);
  gmax_kernel<<<dim3(1024 / 256, BATCH), blk, 0, stream>>>(part, feat);
}
